// mCAM_20882130993286
// MI455X (gfx1250) — matrix operand order repaired and checked by reading, not compiled, not run on hardware
//
#include <hip/hip_runtime.h>
#include <hip/hip_bf16.h>

// mCAM channel-attention for MI455X (gfx1250, wave32, WMMA, async-LDS).
// B=8, C=512, N=4096.  bf16 WMMA (16x16x32) with f32 accumulation.
// Workspace: qb[32M] kb[32M] vb[32M] energy_f32[8M] att_bf16[4M] = ~108MB.

#define BATCH 8
#define CDIM  512
#define NDIM  4096

typedef __attribute__((ext_vector_type(16))) __bf16 v16bf;
typedef __attribute__((ext_vector_type(8)))  float  v8f;
typedef int v4i __attribute__((vector_size(16)));

union FragBF { v16bf v; unsigned int u[8]; };

#if __has_builtin(__builtin_amdgcn_global_load_async_to_lds_b128) && \
    __has_builtin(__builtin_amdgcn_s_wait_asynccnt)
#define USE_ASYNC 1
#define ASYNC_B128(g, l) __builtin_amdgcn_global_load_async_to_lds_b128( \
    (__attribute__((address_space(1))) v4i*)(g),                         \
    (__attribute__((address_space(3))) v4i*)(l), 0, 0)
#define WAITA() __builtin_amdgcn_s_wait_asynccnt(0)
#else
#define USE_ASYNC 0
#define WAITA()
#endif

__device__ __forceinline__ unsigned short f2bf(float f) {
    unsigned int u = __float_as_uint(f);
    unsigned int r = u + 0x7FFFu + ((u >> 16) & 1u);   // round-to-nearest-even
    return (unsigned short)(r >> 16);
}

// A fragment: 16x32 bf16.  lane<16 -> row M=lane, K in {0..7,16..23};
// lane>=16 -> row M=lane-16, K in {8..15,24..31}.  Pairs (K,K+1) contiguous.
__device__ __forceinline__ v16bf load_frag_a(const unsigned short* t, int stride) {
    int lane = threadIdx.x & 31;
    int lm = lane & 15, hs = lane >> 4;
    const unsigned short* r = t + lm * stride;
    FragBF f;
#pragma unroll
    for (int p = 0; p < 8; ++p) {
        int K = 8 * hs + (p & 3) * 2 + 16 * (p >> 2);
        f.u[p] = *(const unsigned int*)(r + K);
    }
    return f.v;
}

// B fragment: 32x16 bf16, LDS staged as [col][K] (K contiguous).
// lane<16 -> col N=lane, K {0..7,16..23} ; lane>=16 -> col N=lane-16, K {8..15,24..31}.
__device__ __forceinline__ v16bf load_frag_b(const unsigned short* t, int stride) {
    int lane = threadIdx.x & 31;
    int nc = lane & 15, hs = lane >> 4;
    const unsigned short* r = t + nc * stride;
    FragBF f;
#pragma unroll
    for (int p = 0; p < 8; ++p) f.u[p] = *(const unsigned int*)(r + 8 * hs + (p & 3) * 2 + 16 * (p >> 2));
    return f.v;
}

#define WMMA_BF16(A, Bf, Cacc) \
    __builtin_amdgcn_wmma_f32_16x16x32_bf16(false, (A), false, (Bf), (short)0, (Cacc), false, false)

#define LDS_STRIDE 40   // 32 elems + pad; 80B rows -> 16B aligned, staggered banks

// Block tile: 128(M) x 128(N), K-step 32.  8 waves: wm=wid>>1 (M/32), wn=wid&1 (N/64).
// Each wave: 2x4 fragments = 8 WMMA per K-step.

// ---------------------------------------------------------------------------
// Kernel 1: QKV projection.  out[d,n] = sum_c W[d,c] x[b,c,n] + bias[d]
// grid: (NDIM/128, CDIM/128, 3*BATCH), block 256.
// ---------------------------------------------------------------------------
__global__ __launch_bounds__(256) void k_qkv(
    const float* __restrict__ x,
    const float* __restrict__ wq, const float* __restrict__ bq,
    const float* __restrict__ wk, const float* __restrict__ bk,
    const float* __restrict__ wv, const float* __restrict__ bv,
    unsigned short* __restrict__ qb, unsigned short* __restrict__ kb,
    unsigned short* __restrict__ vb)
{
    __shared__ unsigned short ldsA[2][128 * LDS_STRIDE];
    __shared__ unsigned short ldsB[2][128 * LDS_STRIDE];

    const int sel = blockIdx.z % 3;
    const int b   = blockIdx.z / 3;
    const float* W  = (sel == 0) ? wq : (sel == 1) ? wk : wv;
    const float* bs = (sel == 0) ? bq : (sel == 1) ? bk : bv;
    unsigned short* out = (sel == 0) ? qb : (sel == 1) ? kb : vb;

    const int n0 = blockIdx.x * 128;
    const int d0 = blockIdx.y * 128;
    const int t  = threadIdx.x;
    const int wid = t >> 5, lane = t & 31;
    const int wm = wid >> 1, wn = wid & 1, hs = lane >> 4;

    v8f acc[2][4];
#pragma unroll
    for (int i = 0; i < 2; ++i)
#pragma unroll
        for (int j = 0; j < 4; ++j) acc[i][j] = (v8f){};

    auto stage = [&](int kt, int bufi) {
        // A tile: W[d0..+128][kt*32..+32] f32 -> bf16
#pragma unroll
        for (int i = 0; i < 4; ++i) {
            int idx = t + i * 256;              // 0..1023 float4s
            int row = idx >> 3;
            int c4  = (idx & 7) * 4;
            float4 f = *(const float4*)(W + (size_t)(d0 + row) * CDIM + kt * 32 + c4);
            unsigned short* dst = &ldsA[bufi][row * LDS_STRIDE + c4];
            dst[0] = f2bf(f.x); dst[1] = f2bf(f.y); dst[2] = f2bf(f.z); dst[3] = f2bf(f.w);
        }
        // B tile: x[b, kt*32..+32, n0..+128] transposed into [n][k], f32 -> bf16
#pragma unroll
        for (int i = 0; i < 4; ++i) {
            int idx = t + i * 256;              // 0..1023 float4s
            int row = idx >> 5;                 // k-row 0..31
            int c4  = (idx & 31) * 4;           // n offset 0..124
            float4 f = *(const float4*)(x + ((size_t)b * CDIM + kt * 32 + row) * NDIM + n0 + c4);
            ldsB[bufi][(c4 + 0) * LDS_STRIDE + row] = f2bf(f.x);
            ldsB[bufi][(c4 + 1) * LDS_STRIDE + row] = f2bf(f.y);
            ldsB[bufi][(c4 + 2) * LDS_STRIDE + row] = f2bf(f.z);
            ldsB[bufi][(c4 + 3) * LDS_STRIDE + row] = f2bf(f.w);
        }
    };

    stage(0, 0);
    for (int kt = 0; kt < CDIM / 32; ++kt) {
        const int cur = kt & 1;
        __syncthreads();
        if (kt + 1 < CDIM / 32) stage(kt + 1, cur ^ 1);
        v16bf a0 = load_frag_a(&ldsA[cur][(wm * 32 + 0)  * LDS_STRIDE], LDS_STRIDE);
        v16bf a1 = load_frag_a(&ldsA[cur][(wm * 32 + 16) * LDS_STRIDE], LDS_STRIDE);
        v16bf bf[4];
#pragma unroll
        for (int j = 0; j < 4; ++j)
            bf[j] = load_frag_b(&ldsB[cur][(wn * 64 + j * 16) * LDS_STRIDE], LDS_STRIDE);
#pragma unroll
        for (int j = 0; j < 4; ++j) {
            acc[0][j] = WMMA_BF16(a0, bf[j], acc[0][j]);
            acc[1][j] = WMMA_BF16(a1, bf[j], acc[1][j]);
        }
    }

#pragma unroll
    for (int fi = 0; fi < 2; ++fi)
#pragma unroll
        for (int fj = 0; fj < 4; ++fj) {
            int col = n0 + wn * 64 + fj * 16 + (lane & 15);
#pragma unroll
            for (int r = 0; r < 8; ++r) {
                int row = d0 + wm * 32 + fi * 16 + 8 * hs + r;
                float vvv = acc[fi][fj][r] + bs[row];
                out[((size_t)b * CDIM + row) * NDIM + col] = f2bf(vvv);
            }
        }
}

// ---------------------------------------------------------------------------
// Kernel 2: energy[b,c,d] = sum_n q[b,c,n] * k[b,d,n]
// grid: (CDIM/128, CDIM/128, BATCH), block 256.  Fully async-staged.
// ---------------------------------------------------------------------------
__global__ __launch_bounds__(256) void k_energy(
    const unsigned short* __restrict__ qb,
    const unsigned short* __restrict__ kb,
    float* __restrict__ energy)
{
    __shared__ unsigned short ldsA[2][128 * LDS_STRIDE];
    __shared__ unsigned short ldsB[2][128 * LDS_STRIDE];

    const int b  = blockIdx.z;
    const int d0 = blockIdx.x * 128;    // columns (d)
    const int c0 = blockIdx.y * 128;    // rows (c)
    const int t  = threadIdx.x;
    const int wid = t >> 5, lane = t & 31;
    const int wm = wid >> 1, wn = wid & 1, hs = lane >> 4;

    const unsigned short* qrow = qb + (size_t)b * CDIM * NDIM;
    const unsigned short* krow = kb + (size_t)b * CDIM * NDIM;

    v8f acc[2][4];
#pragma unroll
    for (int i = 0; i < 2; ++i)
#pragma unroll
        for (int j = 0; j < 4; ++j) acc[i][j] = (v8f){};

    auto stage = [&](int nt, int bufi) {
#pragma unroll
        for (int i = 0; i < 2; ++i) {
            int idx = t + i * 256;              // 0..511 b128s
            int row = idx >> 2;                 // 0..127
            int q4  = (idx & 3) * 8;
            const unsigned short* gA = qrow + (size_t)(c0 + row) * NDIM + nt * 32 + q4;
            const unsigned short* gB = krow + (size_t)(d0 + row) * NDIM + nt * 32 + q4;
            unsigned short* lA = &ldsA[bufi][row * LDS_STRIDE + q4];
            unsigned short* lB = &ldsB[bufi][row * LDS_STRIDE + q4];
#if USE_ASYNC
            ASYNC_B128(gA, lA);
            ASYNC_B128(gB, lB);
#else
            *(uint4*)lA = *(const uint4*)gA;
            *(uint4*)lB = *(const uint4*)gB;
#endif
        }
    };

    stage(0, 0);
    for (int nt = 0; nt < NDIM / 32; ++nt) {
        const int cur = nt & 1;
        WAITA();
        __syncthreads();
        if (nt + 1 < NDIM / 32) stage(nt + 1, cur ^ 1);
        v16bf a0 = load_frag_a(&ldsA[cur][(wm * 32 + 0)  * LDS_STRIDE], LDS_STRIDE);
        v16bf a1 = load_frag_a(&ldsA[cur][(wm * 32 + 16) * LDS_STRIDE], LDS_STRIDE);
        v16bf bf[4];
#pragma unroll
        for (int j = 0; j < 4; ++j)
            bf[j] = load_frag_b(&ldsB[cur][(wn * 64 + j * 16) * LDS_STRIDE], LDS_STRIDE);
#pragma unroll
        for (int j = 0; j < 4; ++j) {
            acc[0][j] = WMMA_BF16(a0, bf[j], acc[0][j]);
            acc[1][j] = WMMA_BF16(a1, bf[j], acc[1][j]);
        }
    }

#pragma unroll
    for (int fi = 0; fi < 2; ++fi)
#pragma unroll
        for (int fj = 0; fj < 4; ++fj) {
            int col = d0 + wn * 64 + fj * 16 + (lane & 15);
#pragma unroll
            for (int r = 0; r < 8; ++r) {
                int row = c0 + wm * 32 + fi * 16 + 8 * hs + r;
                energy[((size_t)b * CDIM + row) * CDIM + col] = acc[fi][fj][r];
            }
        }
}

// ---------------------------------------------------------------------------
// Kernel 3: attention = softmax(rowmax - energy) == exp(emin - e)/sum
// one block (256 thr) per row of 512.  grid: BATCH*CDIM.
// ---------------------------------------------------------------------------
__global__ __launch_bounds__(256) void k_softmax(
    const float* __restrict__ E, unsigned short* __restrict__ A)
{
    __shared__ float red[8];
    const int row = blockIdx.x;
    const float* e = E + (size_t)row * CDIM;
    const int t = threadIdx.x, lane = t & 31, wid = t >> 5;

    float v0 = e[t], v1 = e[t + 256];
    float mn = fminf(v0, v1);
#pragma unroll
    for (int off = 16; off; off >>= 1) mn = fminf(mn, __shfl_xor(mn, off));
    if (lane == 0) red[wid] = mn;
    __syncthreads();
    float m = red[lane & 7];
#pragma unroll
    for (int off = 4; off; off >>= 1) m = fminf(m, __shfl_xor(m, off));
    m = __shfl(m, 0);
    __syncthreads();

    float p0 = __expf(m - v0), p1 = __expf(m - v1);
    float s = p0 + p1;
#pragma unroll
    for (int off = 16; off; off >>= 1) s += __shfl_xor(s, off);
    if (lane == 0) red[wid] = s;
    __syncthreads();
    float ss = red[lane & 7];
#pragma unroll
    for (int off = 4; off; off >>= 1) ss += __shfl_xor(ss, off);
    ss = __shfl(ss, 0);

    float inv = 1.0f / ss;
    unsigned short* a = A + (size_t)row * CDIM;
    a[t]       = f2bf(p0 * inv);
    a[t + 256] = f2bf(p1 * inv);
}

// ---------------------------------------------------------------------------
// Kernel 4: out[b,c,n] = gamma * (sum_d att[b,c,d] v[b,d,n]) + x[b,c,n]
// grid: (NDIM/128, CDIM/128, BATCH), block 256.  A async, B manual transpose.
// ---------------------------------------------------------------------------
__global__ __launch_bounds__(256) void k_out(
    const unsigned short* __restrict__ att,
    const unsigned short* __restrict__ vb,
    const float* __restrict__ x,
    const float* __restrict__ gamma,
    float* __restrict__ out)
{
    __shared__ unsigned short ldsA[2][128 * LDS_STRIDE];
    __shared__ unsigned short ldsB[2][128 * LDS_STRIDE];

    const int b  = blockIdx.z;
    const int n0 = blockIdx.x * 128;
    const int c0 = blockIdx.y * 128;
    const int t  = threadIdx.x;
    const int wid = t >> 5, lane = t & 31;
    const int wm = wid >> 1, wn = wid & 1, hs = lane >> 4;

    const unsigned short* arow = att + (size_t)b * CDIM * CDIM;
    const unsigned short* vrow = vb  + (size_t)b * CDIM * NDIM;

    v8f acc[2][4];
#pragma unroll
    for (int i = 0; i < 2; ++i)
#pragma unroll
        for (int j = 0; j < 4; ++j) acc[i][j] = (v8f){};

    auto stage = [&](int dt, int bufi) {
        // A: att[c0..+128][dt*32..+32] — row-major copy
#pragma unroll
        for (int i = 0; i < 2; ++i) {
            int idx = t + i * 256;
            int row = idx >> 2;
            int q4  = (idx & 3) * 8;
            const unsigned short* g = arow + (size_t)(c0 + row) * CDIM + dt * 32 + q4;
            unsigned short* l = &ldsA[bufi][row * LDS_STRIDE + q4];
#if USE_ASYNC
            ASYNC_B128(g, l);
#else
            *(uint4*)l = *(const uint4*)g;
#endif
        }
        // B: v[dt*32..+32][n0..+128] transposed to [n][d]
#pragma unroll
        for (int i = 0; i < 8; ++i) {
            int idx = t + i * 256;              // 0..2047 uints
            int dr  = idx >> 6;                 // 0..31
            int nc2 = idx & 63;                 // uint index along n
            unsigned int v = *(const unsigned int*)(vrow + (size_t)(dt * 32 + dr) * NDIM + n0 + nc2 * 2);
            ldsB[bufi][(nc2 * 2 + 0) * LDS_STRIDE + dr] = (unsigned short)(v & 0xFFFFu);
            ldsB[bufi][(nc2 * 2 + 1) * LDS_STRIDE + dr] = (unsigned short)(v >> 16);
        }
    };

    stage(0, 0);
    for (int dt = 0; dt < CDIM / 32; ++dt) {
        const int cur = dt & 1;
        WAITA();
        __syncthreads();
        if (dt + 1 < CDIM / 32) stage(dt + 1, cur ^ 1);
        v16bf a0 = load_frag_a(&ldsA[cur][(wm * 32 + 0)  * LDS_STRIDE], LDS_STRIDE);
        v16bf a1 = load_frag_a(&ldsA[cur][(wm * 32 + 16) * LDS_STRIDE], LDS_STRIDE);
        v16bf bf[4];
#pragma unroll
        for (int j = 0; j < 4; ++j)
            bf[j] = load_frag_b(&ldsB[cur][(wn * 64 + j * 16) * LDS_STRIDE], LDS_STRIDE);
#pragma unroll
        for (int j = 0; j < 4; ++j) {
            acc[0][j] = WMMA_BF16(a0, bf[j], acc[0][j]);
            acc[1][j] = WMMA_BF16(a1, bf[j], acc[1][j]);
        }
    }

    const float g = gamma[0];
#pragma unroll
    for (int fi = 0; fi < 2; ++fi)
#pragma unroll
        for (int fj = 0; fj < 4; ++fj) {
            int col = n0 + wn * 64 + fj * 16 + (lane & 15);
#pragma unroll
            for (int r = 0; r < 8; ++r) {
                int row = c0 + wm * 32 + fi * 16 + 8 * hs + r;
                size_t off = ((size_t)b * CDIM + row) * NDIM + col;
                out[off] = g * acc[fi][fj][r] + x[off];
            }
        }
}

// ---------------------------------------------------------------------------
extern "C" void kernel_launch(void* const* d_in, const int* in_sizes, int n_in,
                              void* d_out, int out_size, void* d_ws, size_t ws_size,
                              hipStream_t stream) {
    const float* x  = (const float*)d_in[0];
    const float* wq = (const float*)d_in[1];
    const float* bq = (const float*)d_in[2];
    const float* wk = (const float*)d_in[3];
    const float* bk = (const float*)d_in[4];
    const float* wv = (const float*)d_in[5];
    const float* bv = (const float*)d_in[6];
    const float* gm = (const float*)d_in[7];
    float* out = (float*)d_out;

    const size_t BCN = (size_t)BATCH * CDIM * NDIM;
    const size_t BCC = (size_t)BATCH * CDIM * CDIM;
    unsigned short* qb = (unsigned short*)d_ws;
    unsigned short* kb = qb + BCN;
    unsigned short* vb = kb + BCN;
    float* energy = (float*)(vb + BCN);
    unsigned short* attb = (unsigned short*)(energy + BCC);

    k_qkv<<<dim3(NDIM / 128, CDIM / 128, 3 * BATCH), 256, 0, stream>>>(
        x, wq, bq, wk, bk, wv, bv, qb, kb, vb);
    k_energy<<<dim3(CDIM / 128, CDIM / 128, BATCH), 256, 0, stream>>>(qb, kb, energy);
    k_softmax<<<dim3(BATCH * CDIM), 256, 0, stream>>>(energy, attb);
    k_out<<<dim3(NDIM / 128, CDIM / 128, BATCH), 256, 0, stream>>>(attb, vb, x, gm, out);
}
